// GnnClassifier_70866960384184
// MI455X (gfx1250) — matrix operand order repaired and checked by reading, not compiled, not run on hardware
//
#include <hip/hip_runtime.h>
#include <stdint.h>

#define NNODES 20000
#define NEDGES 320000
#define DIN    2048
#define HD1    512
#define HD2    128
#define HD3    32
#define LN_EPS 1e-5f

typedef __bf16 bf16_t;
typedef __attribute__((ext_vector_type(16))) __bf16 v16bf;
typedef __attribute__((ext_vector_type(4)))  __bf16 v4bf;
typedef __attribute__((ext_vector_type(8)))  float  v8f;
typedef __attribute__((ext_vector_type(4)))  float  fvec4;

// monotone encode of float so unsigned-int atomicMax == float max (handles negatives)
__device__ __forceinline__ unsigned fenc(float f) {
    unsigned u = __float_as_uint(f);
    return (u & 0x80000000u) ? ~u : (u | 0x80000000u);
}
__device__ __forceinline__ float fdec(unsigned u) {
    return (u & 0x80000000u) ? __uint_as_float(u & 0x7fffffffu) : __uint_as_float(~u);
}

// ============================================================================
// WMMA GEMMs.  C[M x Nout] = A[M x K] * W[K x Nout], Wt = bf16 [Nout x K]
// (pre-transposed so each lane's B fragment is two contiguous 8-element runs).
// One wave computes a 16 x (16*NT) strip: one A fragment is reused across NT
// v_wmma_f32_16x16x32_bf16 issues per K-step.
//
// ISA fragment layouts (wave32):
//   A (16x32 bf16): lane<16 holds M=lane, K = {k0+e (e<8), k0+8+e (e>=8)};
//                   lane>=16 same rows with K offset +8  -> two 16B runs.
//   B (32x16 bf16): lane = column, same K split as A (+8 for lanes>=16) -> two 16B runs.
//   C/D (16x16 f32): lane = column, VGPR r -> row r (+8 for lanes>=16).
// ============================================================================

template <int NT>
__global__ __launch_bounds__(32) void k_gemm_bf16in(
    const bf16_t* __restrict__ A, const bf16_t* __restrict__ Wt,
    float* __restrict__ C, int K, int Nout)
{
    const int lane = threadIdx.x;
    const int l16  = lane & 15;
    const int half = lane >> 4;
    const int tm   = blockIdx.x * 16;
    const int tn   = blockIdx.y * (16 * NT);

    const bf16_t* arow = A + (size_t)(tm + l16) * K;

    v8f acc[NT] = {};
    for (int k0 = 0; k0 < K; k0 += 32) {
        union { v16bf v; fvec4 q[2]; } av;
        av.q[0] = *reinterpret_cast<const fvec4*>(arow + k0 + half * 8);
        av.q[1] = *reinterpret_cast<const fvec4*>(arow + k0 + 16 + half * 8);
        #pragma unroll
        for (int t = 0; t < NT; ++t) {
            const bf16_t* wrow = Wt + (size_t)(tn + t * 16 + l16) * K;
            union { v16bf v; fvec4 q[2]; } bv;
            bv.q[0] = *reinterpret_cast<const fvec4*>(wrow + k0 + half * 8);
            bv.q[1] = *reinterpret_cast<const fvec4*>(wrow + k0 + 16 + half * 8);
            acc[t] = __builtin_amdgcn_wmma_f32_16x16x32_bf16(
                false, av.v, false, bv.v, (short)0, acc[t], false, false);
        }
    }
    #pragma unroll
    for (int t = 0; t < NT; ++t)
        #pragma unroll
        for (int r = 0; r < 8; ++r)
            C[(size_t)(tm + half * 8 + r) * Nout + tn + t * 16 + l16] = acc[t][r];
}

// Layer-1 variant: A is f32 (the raw input x); converted in-register with
// native packed f32->bf16 converts (__builtin_convertvector).
template <int NT>
__global__ __launch_bounds__(32) void k_gemm_f32in(
    const float* __restrict__ A, const bf16_t* __restrict__ Wt,
    float* __restrict__ C, int K, int Nout)
{
    const int lane = threadIdx.x;
    const int l16  = lane & 15;
    const int half = lane >> 4;
    const int tm   = blockIdx.x * 16;
    const int tn   = blockIdx.y * (16 * NT);

    const float* arow = A + (size_t)(tm + l16) * K;

    v8f acc[NT] = {};
    for (int k0 = 0; k0 < K; k0 += 32) {
        const int ka0 = k0 + half * 8;
        const int ka1 = k0 + 16 + half * 8;
        fvec4 f0 = *reinterpret_cast<const fvec4*>(arow + ka0);
        fvec4 f1 = *reinterpret_cast<const fvec4*>(arow + ka0 + 4);
        fvec4 f2 = *reinterpret_cast<const fvec4*>(arow + ka1);
        fvec4 f3 = *reinterpret_cast<const fvec4*>(arow + ka1 + 4);
        union { v16bf v; v4bf q[4]; } av;
        av.q[0] = __builtin_convertvector(f0, v4bf);
        av.q[1] = __builtin_convertvector(f1, v4bf);
        av.q[2] = __builtin_convertvector(f2, v4bf);
        av.q[3] = __builtin_convertvector(f3, v4bf);
        #pragma unroll
        for (int t = 0; t < NT; ++t) {
            const bf16_t* wrow = Wt + (size_t)(tn + t * 16 + l16) * K;
            union { v16bf v; fvec4 q[2]; } bv;
            bv.q[0] = *reinterpret_cast<const fvec4*>(wrow + k0 + half * 8);
            bv.q[1] = *reinterpret_cast<const fvec4*>(wrow + k0 + 16 + half * 8);
            acc[t] = __builtin_amdgcn_wmma_f32_16x16x32_bf16(
                false, av.v, false, bv.v, (short)0, acc[t], false, false);
        }
    }
    #pragma unroll
    for (int t = 0; t < NT; ++t)
        #pragma unroll
        for (int r = 0; r < 8; ++r)
            C[(size_t)(tm + half * 8 + r) * Nout + tn + t * 16 + l16] = acc[t][r];
}

// transpose + f32->bf16: Wt[n*K + k] = W[k*Nout + n]
__global__ void k_wt(const float* __restrict__ W, bf16_t* __restrict__ Wt, int K, int Nout) {
    int idx = blockIdx.x * blockDim.x + threadIdx.x;
    if (idx >= K * Nout) return;
    int n = idx / K, k = idx - n * K;
    Wt[idx] = (bf16_t)W[(size_t)k * Nout + n];
}

// ---------------- graph degree ----------------
__global__ void k_deg(const long long* __restrict__ ei, float* __restrict__ deg) {
    int e = blockIdx.x * blockDim.x + threadIdx.x;
    if (e >= NEDGES) return;
    atomicAdd(&deg[(int)ei[NEDGES + e]], 1.0f);
}
__global__ void k_dinv(const float* __restrict__ deg, float* __restrict__ dinv) {
    int i = blockIdx.x * blockDim.x + threadIdx.x;
    if (i >= NNODES) return;
    dinv[i] = rsqrtf(deg[i] + 1.0f);   // +1 self loop; deg>=1 always
}

// ---------------- GCN edge aggregation: agg[dst] += xw[src] * dinv[s]*dinv[d] --
__global__ void k_gcn_edge(const long long* __restrict__ ei, const float* __restrict__ dinv,
                           const float* __restrict__ xw, float* __restrict__ agg, int H) {
    int t  = blockIdx.x * blockDim.x + threadIdx.x;
    int hc = H >> 2;
    int e  = t / hc;
    int f  = (t - e * hc) << 2;
    if (e >= NEDGES) return;
    int s = (int)ei[e], d = (int)ei[NEDGES + e];
    float norm = dinv[s] * dinv[d];
    fvec4 v = *reinterpret_cast<const fvec4*>(xw + (size_t)s * H + f);
    float* o = agg + (size_t)d * H + f;
    atomicAdd(o + 0, v.x * norm);
    atomicAdd(o + 1, v.y * norm);
    atomicAdd(o + 2, v.z * norm);
    atomicAdd(o + 3, v.w * norm);
}

// ---- self-loop + bias + LayerNorm + ReLU (wave per node); emits bf16 h ----
template <int H>
__global__ __launch_bounds__(32) void k_gcn_post(
    const float* __restrict__ xw, const float* __restrict__ dinv,
    const float* __restrict__ agg, const float* __restrict__ bias,
    const float* __restrict__ g, const float* __restrict__ bv,
    bf16_t* __restrict__ hout)
{
    constexpr int C = H / 32;
    int i = blockIdx.x, lane = threadIdx.x;
    float selfn = dinv[i] * dinv[i];
    float vals[C];
    float sum = 0.f;
    #pragma unroll
    for (int j = 0; j < C; ++j) {
        int f = j * 32 + lane;
        float v = agg[(size_t)i * H + f] + xw[(size_t)i * H + f] * selfn + bias[f];
        vals[j] = v; sum += v;
    }
    #pragma unroll
    for (int m = 16; m; m >>= 1) sum += __shfl_xor(sum, m, 32);
    float mu = sum * (1.0f / (float)H);
    float vs = 0.f;
    #pragma unroll
    for (int j = 0; j < C; ++j) { float d0 = vals[j] - mu; vs += d0 * d0; }
    #pragma unroll
    for (int m = 16; m; m >>= 1) vs += __shfl_xor(vs, m, 32);
    float inv = rsqrtf(vs * (1.0f / (float)H) + LN_EPS);
    #pragma unroll
    for (int j = 0; j < C; ++j) {
        int f = j * 32 + lane;
        float o = (vals[j] - mu) * inv * g[f] + bv[f];
        hout[(size_t)i * H + f] = (bf16_t)fmaxf(o, 0.f);
    }
}

// ---------------- GAT ----------------
__global__ __launch_bounds__(32) void k_gat_node1(
    const float* __restrict__ xwg, const float* __restrict__ asrc, const float* __restrict__ adst,
    float* __restrict__ ssrc, float* __restrict__ sdst, float* __restrict__ aself,
    unsigned* __restrict__ amaxu)
{
    int i = blockIdx.x, lane = threadIdx.x;
    float v  = xwg[(size_t)i * HD3 + lane];
    float s1 = v * asrc[lane], s2 = v * adst[lane];
    #pragma unroll
    for (int m = 16; m; m >>= 1) { s1 += __shfl_xor(s1, m, 32); s2 += __shfl_xor(s2, m, 32); }
    if (lane == 0) {
        ssrc[i] = s1; sdst[i] = s2;
        float a = s1 + s2; a = a > 0.f ? a : 0.2f * a;   // self-loop alpha
        aself[i] = a; amaxu[i] = fenc(a);
    }
}

__global__ void k_gat_max(const long long* __restrict__ ei, const float* __restrict__ ssrc,
                          const float* __restrict__ sdst, unsigned* __restrict__ amaxu) {
    int e = blockIdx.x * blockDim.x + threadIdx.x;
    if (e >= NEDGES) return;
    int s = (int)ei[e], d = (int)ei[NEDGES + e];
    float a = ssrc[s] + sdst[d]; a = a > 0.f ? a : 0.2f * a;
    atomicMax(&amaxu[d], fenc(a));
}

__global__ void k_gat_node2(const unsigned* __restrict__ amaxu, const float* __restrict__ aself,
                            float* __restrict__ amaxf, float* __restrict__ denom) {
    int i = blockIdx.x * blockDim.x + threadIdx.x;
    if (i >= NNODES) return;
    float m = fdec(amaxu[i]);
    amaxf[i] = m;
    denom[i] = expf(aself[i] - m);     // self-loop term
}

__global__ void k_gat_denom(const long long* __restrict__ ei, const float* __restrict__ ssrc,
                            const float* __restrict__ sdst, const float* __restrict__ amaxf,
                            float* __restrict__ denom) {
    int e = blockIdx.x * blockDim.x + threadIdx.x;
    if (e >= NEDGES) return;
    int s = (int)ei[e], d = (int)ei[NEDGES + e];
    float a = ssrc[s] + sdst[d]; a = a > 0.f ? a : 0.2f * a;
    atomicAdd(&denom[d], expf(a - amaxf[d]));
}

__global__ void k_gat_msg(const long long* __restrict__ ei, const float* __restrict__ ssrc,
                          const float* __restrict__ sdst, const float* __restrict__ amaxf,
                          const float* __restrict__ denom, const float* __restrict__ xwg,
                          float* __restrict__ gout) {
    int t = blockIdx.x * blockDim.x + threadIdx.x;
    int e = t >> 5, f = t & 31;
    if (e >= NEDGES) return;
    int s = (int)ei[e], d = (int)ei[NEDGES + e];
    float a = ssrc[s] + sdst[d]; a = a > 0.f ? a : 0.2f * a;
    float coef = expf(a - amaxf[d]) / denom[d];
    atomicAdd(&gout[(size_t)d * HD3 + f], xwg[(size_t)s * HD3 + f] * coef);
}

__global__ __launch_bounds__(32) void k_gat_final(
    const float* __restrict__ gout, const float* __restrict__ xwg,
    const float* __restrict__ aself, const float* __restrict__ amaxf,
    const float* __restrict__ denom, const float* __restrict__ bg,
    float* __restrict__ pool)
{
    int i = blockIdx.x, lane = threadIdx.x;
    float coef = expf(aself[i] - amaxf[i]) / denom[i];
    float h = gout[(size_t)i * HD3 + lane] + xwg[(size_t)i * HD3 + lane] * coef + bg[lane];
    h = fmaxf(h, 0.f);
    atomicAdd(&pool[lane], h);
}

__global__ void k_fc(const float* __restrict__ pool, const float* __restrict__ fcw,
                     const float* __restrict__ fcb, float* __restrict__ out) {
    int j = threadIdx.x;
    if (j >= 2) return;
    float s = fcb[j];
    for (int f = 0; f < HD3; ++f)
        s += (pool[f] * (1.0f / (float)NNODES)) * fcw[f * 2 + j];
    out[j] = s;
}

// ---------------- host-side GCN layer driver (layers 2,3: bf16 activations) --
template <int H, int NT>
static void run_gcn_bf16(const bf16_t* Ain, int K, const float* W, const float* bias,
                         const float* g, const float* bv, bf16_t* hout,
                         const long long* ei, const float* dinv,
                         bf16_t* wT, float* bufA, float* bufB, hipStream_t stream) {
    int kw = K * H;
    k_wt<<<(kw + 255) / 256, 256, 0, stream>>>(W, wT, K, H);
    dim3 gg(NNODES / 16, H / (16 * NT));
    k_gemm_bf16in<NT><<<gg, 32, 0, stream>>>(Ain, wT, bufA, K, H);
    hipMemsetAsync(bufB, 0, (size_t)NNODES * H * sizeof(float), stream);
    int tot = NEDGES * (H / 4);
    k_gcn_edge<<<(tot + 255) / 256, 256, 0, stream>>>(ei, dinv, bufA, bufB, H);
    k_gcn_post<H><<<NNODES, 32, 0, stream>>>(bufA, dinv, bufB, bias, g, bv, hout);
}

extern "C" void kernel_launch(void* const* d_in, const int* in_sizes, int n_in,
                              void* d_out, int out_size, void* d_ws, size_t ws_size,
                              hipStream_t stream) {
    const float*     x    = (const float*)d_in[0];
    const long long* ei   = (const long long*)d_in[1];
    const float*     W1   = (const float*)d_in[2];
    const float*     b1   = (const float*)d_in[3];
    const float*     g1   = (const float*)d_in[4];
    const float*     bb1  = (const float*)d_in[5];
    const float*     W2   = (const float*)d_in[6];
    const float*     b2   = (const float*)d_in[7];
    const float*     g2   = (const float*)d_in[8];
    const float*     bb2  = (const float*)d_in[9];
    const float*     W3   = (const float*)d_in[10];
    const float*     b3   = (const float*)d_in[11];
    const float*     g3   = (const float*)d_in[12];
    const float*     bb3  = (const float*)d_in[13];
    const float*     Wg   = (const float*)d_in[14];
    const float*     asrc = (const float*)d_in[15];
    const float*     adst = (const float*)d_in[16];
    const float*     bg   = (const float*)d_in[17];
    const float*     fcw  = (const float*)d_in[18];
    const float*     fcb  = (const float*)d_in[19];
    float*           out  = (float*)d_out;

    char* cur = (char*)d_ws;
    auto walloc = [&](size_t bytes) -> void* {
        void* p = (void*)cur;
        cur += (bytes + 255) & ~(size_t)255;
        return p;
    };
    bf16_t*   wT    = (bf16_t*)walloc((size_t)HD1 * DIN * sizeof(bf16_t));    // 2 MB (largest Wt)
    float*    bufA  = (float*)walloc((size_t)NNODES * HD1 * sizeof(float));   // xw (f32)
    float*    bufB  = (float*)walloc((size_t)NNODES * HD1 * sizeof(float));   // agg (f32)
    bf16_t*   hbf   = (bf16_t*)walloc((size_t)NNODES * HD1 * sizeof(bf16_t)); // h (bf16)
    float*    deg   = (float*)walloc(NNODES * sizeof(float));
    float*    dinv  = (float*)walloc(NNODES * sizeof(float));
    float*    ssrc  = (float*)walloc(NNODES * sizeof(float));
    float*    sdst  = (float*)walloc(NNODES * sizeof(float));
    float*    aself = (float*)walloc(NNODES * sizeof(float));
    unsigned* amaxu = (unsigned*)walloc(NNODES * sizeof(unsigned));
    float*    amaxf = (float*)walloc(NNODES * sizeof(float));
    float*    denom = (float*)walloc(NNODES * sizeof(float));
    float*    pool  = (float*)walloc(HD3 * sizeof(float));

    // degrees (shared by all 3 GCN layers)
    hipMemsetAsync(deg, 0, NNODES * sizeof(float), stream);
    k_deg<<<(NEDGES + 255) / 256, 256, 0, stream>>>(ei, deg);
    k_dinv<<<(NNODES + 255) / 256, 256, 0, stream>>>(deg, dinv);

    // ---- layer 1: f32 x input, in-register packed converts ----
    k_wt<<<(DIN * HD1 + 255) / 256, 256, 0, stream>>>(W1, wT, DIN, HD1);
    k_gemm_f32in<4><<<dim3(NNODES / 16, HD1 / 64), 32, 0, stream>>>(x, wT, bufA, DIN, HD1);
    hipMemsetAsync(bufB, 0, (size_t)NNODES * HD1 * sizeof(float), stream);
    k_gcn_edge<<<(NEDGES * (HD1 / 4) + 255) / 256, 256, 0, stream>>>(ei, dinv, bufA, bufB, HD1);
    k_gcn_post<HD1><<<NNODES, 32, 0, stream>>>(bufA, dinv, bufB, b1, g1, bb1, hbf);

    // ---- layers 2,3: bf16 activations straight into WMMA ----
    // (hbf overwrite by k_gcn_post is safe: the GEMM that read it has completed)
    run_gcn_bf16<HD2, 4>(hbf, HD1, W2, b2, g2, bb2, hbf, ei, dinv, wT, bufA, bufB, stream);
    run_gcn_bf16<HD3, 2>(hbf, HD2, W3, b3, g3, bb3, hbf, ei, dinv, wT, bufA, bufB, stream);

    // ---- GAT layer ----
    k_wt<<<(HD3 * HD3 + 255) / 256, 256, 0, stream>>>(Wg, wT, HD3, HD3);
    k_gemm_bf16in<2><<<dim3(NNODES / 16, 1), 32, 0, stream>>>(hbf, wT, bufA, HD3, HD3);
    k_gat_node1<<<NNODES, 32, 0, stream>>>(bufA, asrc, adst, ssrc, sdst, aself, amaxu);
    k_gat_max<<<(NEDGES + 255) / 256, 256, 0, stream>>>(ei, ssrc, sdst, amaxu);
    k_gat_node2<<<(NNODES + 255) / 256, 256, 0, stream>>>(amaxu, aself, amaxf, denom);
    k_gat_denom<<<(NEDGES + 255) / 256, 256, 0, stream>>>(ei, ssrc, sdst, amaxf, denom);
    hipMemsetAsync(bufB, 0, (size_t)NNODES * HD3 * sizeof(float), stream);
    k_gat_msg<<<(NEDGES * 32 + 255) / 256, 256, 0, stream>>>(ei, ssrc, sdst, amaxf, denom, bufA, bufB);
    hipMemsetAsync(pool, 0, HD3 * sizeof(float), stream);
    k_gat_final<<<NNODES, 32, 0, stream>>>(bufB, bufA, aself, amaxf, denom, bg, pool);
    k_fc<<<1, 32, 0, stream>>>(pool, fcw, fcb, out);
}
